// VQ_83227876262352
// MI455X (gfx1250) — compile-verified, matrix operand order checked by reading, not run on hardware
//
#include <hip/hip_runtime.h>
#include <hip/hip_bf16.h>
#include <stdint.h>

#define D      256
#define NCODES 4096
#define NROWS  16384
#define WPB    4   // waves (M-tiles) per block

typedef __attribute__((ext_vector_type(16))) __bf16 v16bf;
typedef __attribute__((ext_vector_type(8)))  __bf16 v8bf;
typedef __attribute__((ext_vector_type(8)))  float  v8f;

union V16 { v16bf v; v8bf h[2]; };

__device__ inline __bf16 f2bf(float f) {
  uint32_t u = __builtin_bit_cast(uint32_t, f);
  uint32_t r = u + 0x7FFFu + ((u >> 16) & 1u);   // round-to-nearest-even
  uint16_t h = (uint16_t)(r >> 16);
  return __builtin_bit_cast(__bf16, h);
}

// ---- codebook: f32 -> bf16, plus c2[j] = sum(c^2) in f32 -------------------
__global__ void cb_prep(const float* __restrict__ w, __bf16* __restrict__ cb16,
                        float* __restrict__ c2) {
  int j = blockIdx.x, t = threadIdx.x;
  float x = w[(size_t)j * D + t];
  cb16[(size_t)j * D + t] = f2bf(x);
  __shared__ float red[256];
  red[t] = x * x;
  __syncthreads();
  for (int s = 128; s > 0; s >>= 1) { if (t < s) red[t] += red[t + s]; __syncthreads(); }
  if (t == 0) c2[j] = red[0];
}

// ---- v: f32 -> bf16, plus per-row mask any(v != 1/256) ---------------------
__global__ void v_prep(const float* __restrict__ v, __bf16* __restrict__ vb,
                       int* __restrict__ mask) {
  int m = blockIdx.x, t = threadIdx.x;
  float x = v[(size_t)m * D + t];
  vb[(size_t)m * D + t] = f2bf(x);
  __shared__ int flag;
  if (t == 0) flag = 0;
  __syncthreads();
  if (x != (1.0f / 256.0f)) atomicOr(&flag, 1);
  __syncthreads();
  if (t == 0) mask[m] = flag;
}

// ---- main: bf16 WMMA scores + fused argmin ---------------------------------
// Each wave owns one 16-row M-tile; per iteration it processes TWO 16-code
// n-tiles with TWO K-split accumulators each -> 4 independent WMMA chains of
// length 4 (16 v_wmma per loop body) to keep the matrix pipe fed.
__global__ void __launch_bounds__(WPB * 32)
vq_argmin(const __bf16* __restrict__ vb, const __bf16* __restrict__ cb16,
          const float* __restrict__ c2, const int* __restrict__ mask,
          float* __restrict__ idx_out) {
  const int lane  = threadIdx.x & 31;
  const int wave  = threadIdx.x >> 5;
  const int mtile = blockIdx.x * WPB + wave;
  const int rowb  = mtile * 16;
  const int hi    = lane >> 4;        // lane half: 0 or 1
  const int l16   = lane & 15;
  const int koff  = hi * 8;           // K sub-chunk per CDNA5 16-bit A/B layout

  // Preload this wave's A tile (16 rows x 256 K) into VGPRs: 8 x v16bf
  V16 a[8];
  const __bf16* arow = vb + (size_t)(rowb + l16) * D;
  #pragma unroll
  for (int kc = 0; kc < 8; ++kc) {
    const v8bf* p = (const v8bf*)(arow + kc * 32 + koff);
    a[kc].h[0] = p[0];        // K = kc*32 + koff .. +7
    a[kc].h[1] = p[2];        // K = kc*32 + koff + 16 .. +23
  }

  float bmin[8];
  int   bidx[8];
  #pragma unroll
  for (int r = 0; r < 8; ++r) { bmin[r] = 3.4e38f; bidx[r] = 1; }

  for (int nt = 0; nt < NCODES / 16; nt += 2) {
    const int n0 = nt * 16 + l16;            // this lane's code column, tile 0
    const int n1 = n0 + 16;                  // tile 1 (always > n0)
    const __bf16* brow0 = cb16 + (size_t)n0 * D;
    const __bf16* brow1 = cb16 + (size_t)n1 * D;

    v8f acc0a = {}, acc0b = {}, acc1a = {}, acc1b = {};
    #pragma unroll
    for (int kc = 0; kc < 8; kc += 2) {
      V16 b0e, b0o, b1e, b1o;
      const v8bf* p0e = (const v8bf*)(brow0 + kc * 32 + koff);
      const v8bf* p0o = (const v8bf*)(brow0 + (kc + 1) * 32 + koff);
      const v8bf* p1e = (const v8bf*)(brow1 + kc * 32 + koff);
      const v8bf* p1o = (const v8bf*)(brow1 + (kc + 1) * 32 + koff);
      b0e.h[0] = p0e[0]; b0e.h[1] = p0e[2];
      b0o.h[0] = p0o[0]; b0o.h[1] = p0o[2];
      b1e.h[0] = p1e[0]; b1e.h[1] = p1e[2];
      b1o.h[0] = p1o[0]; b1o.h[1] = p1o[2];
      acc0a = __builtin_amdgcn_wmma_f32_16x16x32_bf16(
          false, a[kc].v,     false, b0e.v, (short)0, acc0a, false, false);
      acc0b = __builtin_amdgcn_wmma_f32_16x16x32_bf16(
          false, a[kc + 1].v, false, b0o.v, (short)0, acc0b, false, false);
      acc1a = __builtin_amdgcn_wmma_f32_16x16x32_bf16(
          false, a[kc].v,     false, b1e.v, (short)0, acc1a, false, false);
      acc1b = __builtin_amdgcn_wmma_f32_16x16x32_bf16(
          false, a[kc + 1].v, false, b1o.v, (short)0, acc1b, false, false);
    }

    const float c20 = c2[n0];
    const float c21 = c2[n1];
    // n scans strictly increasing per lane, so strict '<' keeps the first
    // occurrence of the minimum (jnp.argmin semantics) with no tiebreak cost.
    #pragma unroll
    for (int r = 0; r < 8; ++r) {
      float d0 = fmaf(-2.0f, acc0a[r] + acc0b[r], c20);
      if (n0 == 0) d0 = 3.5e38f;               // reference searches codes 1..4095
      if (d0 < bmin[r]) { bmin[r] = d0; bidx[r] = n0; }
      float d1 = fmaf(-2.0f, acc1a[r] + acc1b[r], c21);
      if (d1 < bmin[r]) { bmin[r] = d1; bidx[r] = n1; }
    }
  }

  // argmin across the 16 lanes of each half (rows r / r+8); index tiebreak
  // needed here since different lanes hold different code classes.
  #pragma unroll
  for (int r = 0; r < 8; ++r) {
    #pragma unroll
    for (int off = 8; off; off >>= 1) {
      float om = __shfl_xor(bmin[r], off, 32);
      int   oi = __shfl_xor(bidx[r], off, 32);
      if (om < bmin[r] || (om == bmin[r] && oi < bidx[r])) {
        bmin[r] = om; bidx[r] = oi;
      }
    }
  }

  if (l16 == 0) {
    #pragma unroll
    for (int r = 0; r < 8; ++r) {
      int m  = rowb + r + hi * 8;
      int id = mask[m] ? bidx[r] : 0;
      idx_out[m] = (float)id;
    }
  }
}

// ---- gather out = vectors[idx]; per-row squared-error partial --------------
__global__ void gather_loss(const float* __restrict__ w, const float* __restrict__ v,
                            const float* __restrict__ idx_f, float* __restrict__ out,
                            float* __restrict__ partial) {
  int m = blockIdx.x, t = threadIdx.x;
  int id = (int)idx_f[m];
  float x = w[(size_t)id * D + t];
  out[(size_t)m * D + t] = x;
  float dd = x - v[(size_t)m * D + t];
  __shared__ float red[256];
  red[t] = dd * dd;
  __syncthreads();
  for (int s = 128; s > 0; s >>= 1) { if (t < s) red[t] += red[t + s]; __syncthreads(); }
  if (t == 0) partial[m] = red[0];
}

// ---- deterministic final reduction: loss + used ----------------------------
__global__ void finalize(const float* __restrict__ partial, float* __restrict__ loss,
                         float* __restrict__ used) {
  int t = threadIdx.x;  // 256
  float s = 0.0f;
  for (int i = t; i < NROWS; i += 256) s += partial[i];
  __shared__ float red[256];
  red[t] = s;
  __syncthreads();
  for (int st = 128; st > 0; st >>= 1) { if (t < st) red[t] += red[t + st]; __syncthreads(); }
  if (t == 0) {
    loss[0] = red[0] / (float)((size_t)NROWS * D);
    used[0] = 0.0f;
  }
}

extern "C" void kernel_launch(void* const* d_in, const int* in_sizes, int n_in,
                              void* d_out, int out_size, void* d_ws, size_t ws_size,
                              hipStream_t stream) {
  const float* v = (const float*)d_in[0];   // (8,2048,256)
  const float* w = (const float*)d_in[1];   // (4096,256)

  float* out     = (float*)d_out;                  // 16384*256
  float* idx_out = out + (size_t)NROWS * D;        // 16384 (as float)
  float* loss    = idx_out + NROWS;                // 1
  float* used    = loss + 1;                       // 1

  char* ws = (char*)d_ws;
  __bf16* cb16    = (__bf16*)ws;  ws += (size_t)NCODES * D * sizeof(__bf16);
  __bf16* vb      = (__bf16*)ws;  ws += (size_t)NROWS  * D * sizeof(__bf16);
  float*  c2      = (float*)ws;   ws += (size_t)NCODES * sizeof(float);
  int*    mask    = (int*)ws;     ws += (size_t)NROWS  * sizeof(int);
  float*  partial = (float*)ws;   // NROWS floats

  cb_prep<<<NCODES, 256, 0, stream>>>(w, cb16, c2);
  v_prep<<<NROWS, 256, 0, stream>>>(v, vb, mask);
  vq_argmin<<<(NROWS / 16) / WPB, WPB * 32, 0, stream>>>(vb, cb16, c2, mask, idx_out);
  gather_loss<<<NROWS, 256, 0, stream>>>(w, v, idx_out, out, partial);
  finalize<<<1, 256, 0, stream>>>(partial, loss, used);
}
